// Decoder_8014408974552
// MI455X (gfx1250) — matrix operand order repaired and checked by reading, not compiled, not run on hardware
//
#include <hip/hip_runtime.h>
#include <hip/hip_bf16.h>

// ---------------------------------------------------------------------------
// GRU decoder, persistent-hidden formulation for MI455X (gfx1250).
// Each wave owns 32 batch rows (two 16-row WMMA M-tiles) for all 30 steps;
// hidden lives in LDS (f16, double buffered). W_hh chunk strips are staged
// into LDS once per block with async global->LDS copies (ASYNCcnt), double
// buffered across chunks so the copy overlaps WMMA compute and both waves
// share one L2 fetch. All GEMMs use v_wmma_f32_16x16x32_f16.
// ---------------------------------------------------------------------------

typedef _Float16 v8h  __attribute__((ext_vector_type(8)));
typedef _Float16 v16h __attribute__((ext_vector_type(16)));
typedef float    v8f  __attribute__((ext_vector_type(8)));
typedef __attribute__((__vector_size__(4 * sizeof(int)))) int v4i;

#define B_TOT   32768
#define T_STEPS 30
#define HDIM    384
#define XDIM    128
#define MT      64            // batch rows per block (2 waves x 32 rows)
#define HS      392           // h row stride in halves (384 + 8 pad, 16B mult)
#define D1S     72            // decode-l1 row stride in halves (64 + 8 pad)
#define STGS    392           // staged weight row stride in halves
#define STGROWS 48            // 3 gates x 16 rows per chunk

#if defined(__HIP_DEVICE_COMPILE__) && \
    __has_builtin(__builtin_amdgcn_global_load_async_to_lds_b128)
#define HAVE_ASYNC 1
#else
#define HAVE_ASYNC 0
#endif

// f16 weight scratch layout in d_ws (element offsets, 16B aligned)
#define OFF_WIH  0                               // 1152*128
#define OFF_WHH  (1152*128)                      // 1152*384
#define OFF_WD1  (OFF_WHH + 1152*384)            // 64*384
#define OFF_WD2  (OFF_WD1 + 64*384)              // 16*64 (rows 3..15 zero)

__device__ __forceinline__ v16h cat8(v8h lo, v8h hi) {
  return __builtin_shufflevector(lo, hi, 0,1,2,3,4,5,6,7,8,9,10,11,12,13,14,15);
}
__device__ __forceinline__ v16h loadB16(const _Float16* p) {
  const v8h* q = (const v8h*)p;
  return cat8(q[0], q[1]);
}
__device__ __forceinline__ v16h loadA16(const _Float16* row, int khalf) {
  v8h lo = *(const v8h*)(row + 2 * khalf);
  v8h hi = *(const v8h*)(row + 2 * khalf + 8);
  return cat8(lo, hi);
}
__device__ __forceinline__ v8f wmma(v16h a, v16h b, v8f c) {
  return __builtin_amdgcn_wmma_f32_16x16x32_f16(false, a, false, b,
                                                (short)0, c, false, false);
}
__device__ __forceinline__ float sigmoid_f(float x) {
  return 1.0f / (1.0f + __expf(-x));
}
__device__ __forceinline__ float tanh_f(float x) {
  x = fminf(fmaxf(x, -15.0f), 15.0f);
  float e = __expf(2.0f * x);
  return (e - 1.0f) / (e + 1.0f);
}
__device__ __forceinline__ void wait_async0() {
#if HAVE_ASYNC
#if __has_builtin(__builtin_amdgcn_s_wait_asynccnt)
  __builtin_amdgcn_s_wait_asynccnt(0);
#else
  asm volatile("s_wait_asynccnt 0x0" ::: "memory");
#endif
#endif
}

#if HAVE_ASYNC
// Stage the three W_hh strips (rows {0,384,768}+jn*16 .. +15) for chunk jn
// into LDS: 48 rows x 384 halves = 36 KB, as 2304 16B segments spread over
// 64 threads x 36 async b128 issues (each issue moves 16B per lane).
__device__ __forceinline__ void stage_whh(const _Float16* __restrict__ whh,
                                          _Float16* dst, int jn, int tid) {
#pragma unroll
  for (int q = 0; q < 36; ++q) {
    int e   = tid + q * 64;        // 0..2303
    int row = e / 48;              // 0..47  (g*16 + r)
    int seg = e % 48;              // 16B segment within the 384-half row
    int g   = row >> 4;
    int r   = row & 15;
    const _Float16* gsrc = whh + (g * 384 + jn * 16 + r) * HDIM + seg * 8;
    _Float16*       ldst = dst + row * STGS + seg * 8;
    __builtin_amdgcn_global_load_async_to_lds_b128(
        (__attribute__((address_space(1))) v4i*)gsrc,
        (__attribute__((address_space(3))) v4i*)ldst, 0, 0);
  }
}
#endif

// ---------------------------------------------------------------------------
// Weight conversion: f32 -> f16 into d_ws (Wd2 zero-padded 3 -> 16 rows).
// ---------------------------------------------------------------------------
__global__ void convert_weights_kernel(const float* __restrict__ W_ih,
                                       const float* __restrict__ W_hh,
                                       const float* __restrict__ Wd1,
                                       const float* __restrict__ Wd2,
                                       _Float16* __restrict__ ws) {
  int i = blockIdx.x * blockDim.x + threadIdx.x;
  if (i < 1152 * 128) ws[OFF_WIH + i] = (_Float16)W_ih[i];
  if (i < 1152 * 384) ws[OFF_WHH + i] = (_Float16)W_hh[i];
  if (i < 64 * 384)   ws[OFF_WD1 + i] = (_Float16)Wd1[i];
  if (i < 16 * 64) {
    int r = i >> 6, c = i & 63;
    ws[OFF_WD2 + i] = (r < 3) ? (_Float16)Wd2[r * 64 + c] : (_Float16)0.0f;
  }
}

// ---------------------------------------------------------------------------
// Main persistent-recurrence kernel. 64 threads (2 waves x 32 rows), 512 blocks.
// ---------------------------------------------------------------------------
__global__ __launch_bounds__(64, 1)
void decoder_kernel(const float* __restrict__ init_hidden,  // [B,H]
                    const float* __restrict__ plan,         // [B,T,3]
                    const float* __restrict__ gate,         // [B,1]
                    const float* __restrict__ init_state,   // [B,3]
                    const float* __restrict__ Wp,  const float* __restrict__ bp,
                    const float* __restrict__ Ws,  const float* __restrict__ bs,
                    const float* __restrict__ b_ih, const float* __restrict__ b_hh,
                    const float* __restrict__ bd1, const float* __restrict__ bd2,
                    const _Float16* __restrict__ wih,   // [1152,128] f16
                    const _Float16* __restrict__ whh,   // [1152,384] f16
                    const _Float16* __restrict__ wd1,   // [64,384]  f16
                    const _Float16* __restrict__ wd2p,  // [16,64]   f16 padded
                    float* __restrict__ out)            // [B,T,3]
{
  __shared__ __align__(16) _Float16 hbuf[2][MT * HS];   // hidden f16 ping/pong
#if HAVE_ASYNC
  __shared__ __align__(16) _Float16 stg[2][STGROWS * STGS]; // W_hh chunk strips
#endif
  __shared__ __align__(16) _Float16 d1buf[MT * D1S];    // elu(h@Wd1^T) f16
  __shared__ __align__(16) float    state_s[MT * 4];    // running state f32
  __shared__ __align__(16) float    wsp[1024];          // Ws|Wp|bs|bp f32
  __shared__ __align__(16) float    gbias[4 * HDIM];    // br|bz|bni|bnh f32
  __shared__ __align__(16) float    dbias[80];          // bd1[64] | bd2[3]

  const int tid   = threadIdx.x;
  const int lane  = tid & 31;
  const int wave  = tid >> 5;
  const int l16   = lane & 15;
  const int hi16  = lane >> 4;        // 0 or 1
  const int khalf = hi16 * 8;         // A-fragment half select
  const int ksel  = hi16 * 16;        // B-fragment half select
  const int wrow  = wave * 32;        // this wave's first local row (32 rows)
  const int bbase = blockIdx.x * MT;  // first global batch row of block

  // ---- stage small f32 params ----
  for (int i = tid; i < 384; i += 64) { wsp[i] = Ws[i]; wsp[384 + i] = Wp[i]; }
  for (int i = tid; i < 128; i += 64) { wsp[768 + i] = bs[i]; wsp[896 + i] = bp[i]; }
  for (int i = tid; i < HDIM; i += 64) {
    gbias[i]            = b_ih[i]       + b_hh[i];        // r
    gbias[HDIM + i]     = b_ih[384 + i] + b_hh[384 + i];  // z
    gbias[2 * HDIM + i] = b_ih[768 + i];                  // n (input part)
    gbias[3 * HDIM + i] = b_hh[768 + i];                  // n (hidden part)
  }
  for (int i = tid; i < 64; i += 64) dbias[i] = bd1[i];
  if (tid < 3) dbias[64 + tid] = bd2[tid];

  // ---- init hidden (f32 -> f16 LDS) and state (wave-local rows) ----
  for (int m = 0; m < 32; ++m) {
    int gm = bbase + wrow + m;
    for (int c = lane; c < HDIM; c += 32)
      hbuf[0][(wrow + m) * HS + c] = (_Float16)init_hidden[gm * HDIM + c];
  }
  for (int e = lane; e < 96; e += 32) {
    int m = e / 3, c = e % 3;
    state_s[(wrow + m) * 4 + c] = init_state[(bbase + wrow + m) * 3 + c];
  }
  const float g0 = gate[bbase + wrow + l16];        // gate, tile-0 A-row
  const float g1 = gate[bbase + wrow + 16 + l16];   // gate, tile-1 A-row
  __syncthreads();

  int ping = 0;
  for (int t = 0; t < T_STEPS; ++t) {
    const _Float16* hp = &hbuf[ping][0];
    _Float16*       hq = &hbuf[1 - ping][0];

#if HAVE_ASYNC
    stage_whh(whh, &stg[0][0], 0, tid);   // pre-stage chunk 0 of this step
#endif

    // ---- x = state@Ws^T + bs + gate*(plan_t@Wp^T + bp) as register A-frags
    //      (two 16x128 tiles per wave; lane rows: wrow+l16 and wrow+16+l16) ----
    const int r0 = wrow + l16, r1 = wrow + 16 + l16;
    const float s00 = state_s[r0*4+0], s01 = state_s[r0*4+1], s02 = state_s[r0*4+2];
    const float s10 = state_s[r1*4+0], s11 = state_s[r1*4+1], s12 = state_s[r1*4+2];
    const float* pl0 = &plan[(bbase + r0) * (T_STEPS * 3) + t * 3];
    const float* pl1 = &plan[(bbase + r1) * (T_STEPS * 3) + t * 3];
    const float p00 = pl0[0], p01 = pl0[1], p02 = pl0[2];
    const float p10 = pl1[0], p11 = pl1[1], p12 = pl1[2];

    v16h xa0[4], xa1[4];
#pragma unroll
    for (int f = 0; f < 4; ++f) {
#pragma unroll
      for (int i = 0; i < 16; ++i) {
        int koff = 2 * khalf + i;
        int c = 32 * f + koff;
        float wc0 = wsp[c*3+0], wc1 = wsp[c*3+1], wc2 = wsp[c*3+2];
        float vp0 = wsp[384+c*3+0], vp1 = wsp[384+c*3+1], vp2 = wsp[384+c*3+2];
        float cbs = wsp[768 + c], cbp = wsp[896 + c];
        xa0[f][i] = (_Float16)((wc0*s00 + wc1*s01 + wc2*s02 + cbs) +
                               g0 * (vp0*p00 + vp1*p01 + vp2*p02 + cbp));
        xa1[f][i] = (_Float16)((wc0*s10 + wc1*s11 + wc2*s12 + cbs) +
                               g1 * (vp0*p10 + vp1*p11 + vp2*p12 + cbp));
      }
    }

    // ---- GRU: 24 chunks of 16 hidden units; B fragments reused by 2 tiles ----
    for (int jc = 0; jc < 24; ++jc) {
      const int j = jc * 16 + l16;        // hidden column for this lane
#if HAVE_ASYNC
      wait_async0();                       // my staged copies for buf[jc&1] done
      __syncthreads();                     // all waves done staging + done
                                           // reading buf[(jc+1)&1] last chunk
      if (jc < 23) stage_whh(whh, &stg[(jc + 1) & 1][0], jc + 1, tid);
      const _Float16* sb = &stg[jc & 1][0];
#endif
      v8f ar0 = {}, az0 = {}, ani0 = {}, anh0 = {};
      v8f ar1 = {}, az1 = {}, ani1 = {}, anh1 = {};

      // x part: K = 128 (r, z, n_i) -- direct from L2
#pragma unroll
      for (int kb4 = 0; kb4 < 4; ++kb4) {
        const int kb = kb4 * 32;
        v16h br_ = loadB16(wih + (      j) * XDIM + kb + ksel);
        v16h bz_ = loadB16(wih + (384 + j) * XDIM + kb + ksel);
        v16h bn_ = loadB16(wih + (768 + j) * XDIM + kb + ksel);
        ar0  = wmma(xa0[kb4], br_, ar0);  ar1  = wmma(xa1[kb4], br_, ar1);
        az0  = wmma(xa0[kb4], bz_, az0);  az1  = wmma(xa1[kb4], bz_, az1);
        ani0 = wmma(xa0[kb4], bn_, ani0); ani1 = wmma(xa1[kb4], bn_, ani1);
      }
      // h part: K = 384 (r, z, n_h) -- staged in LDS (shared by both waves)
#pragma unroll
      for (int kb12 = 0; kb12 < 12; ++kb12) {
        const int kb = kb12 * 32;
        v16h a0 = loadA16(hp + (wrow +      l16) * HS + kb, khalf);
        v16h a1 = loadA16(hp + (wrow + 16 + l16) * HS + kb, khalf);
#if HAVE_ASYNC
        v16h br_ = loadB16(sb + (     l16) * STGS + kb + ksel);
        v16h bz_ = loadB16(sb + (16 + l16) * STGS + kb + ksel);
        v16h bn_ = loadB16(sb + (32 + l16) * STGS + kb + ksel);
#else
        v16h br_ = loadB16(whh + (      j) * HDIM + kb + ksel);
        v16h bz_ = loadB16(whh + (384 + j) * HDIM + kb + ksel);
        v16h bn_ = loadB16(whh + (768 + j) * HDIM + kb + ksel);
#endif
        ar0  = wmma(a0, br_, ar0);  ar1  = wmma(a1, br_, ar1);
        az0  = wmma(a0, bz_, az0);  az1  = wmma(a1, bz_, az1);
        anh0 = wmma(a0, bn_, anh0); anh1 = wmma(a1, bn_, anh1);
      }

      // gate math + hidden update (wave-local rows/cols)
      const float br  = gbias[j];
      const float bz  = gbias[HDIM + j];
      const float bni = gbias[2 * HDIM + j];
      const float bnh = gbias[3 * HDIM + j];
#pragma unroll
      for (int v = 0; v < 8; ++v) {
        const int M0 = wrow + v + hi16 * 8;
        const int M1 = M0 + 16;
        float r, z, nn, ho;
        r  = sigmoid_f(ar0[v] + br);
        z  = sigmoid_f(az0[v] + bz);
        nn = tanh_f(ani0[v] + bni + r * (anh0[v] + bnh));
        ho = (float)hp[M0 * HS + j];
        hq[M0 * HS + j] = (_Float16)((1.0f - z) * nn + z * ho);
        r  = sigmoid_f(ar1[v] + br);
        z  = sigmoid_f(az1[v] + bz);
        nn = tanh_f(ani1[v] + bni + r * (anh1[v] + bnh));
        ho = (float)hp[M1 * HS + j];
        hq[M1 * HS + j] = (_Float16)((1.0f - z) * nn + z * ho);
      }
    }
    __syncthreads();

    // ---- decode layer 1: elu(h_new @ Wd1^T + bd1) -> d1buf f16 ----
#pragma unroll
    for (int nc = 0; nc < 4; ++nc) {
      const int n1 = nc * 16 + l16;
      v8f acc0 = {}, acc1 = {};
#pragma unroll
      for (int kb12 = 0; kb12 < 12; ++kb12) {
        const int kb = kb12 * 32;
        v16h a0 = loadA16(hq + (wrow +      l16) * HS + kb, khalf);
        v16h a1 = loadA16(hq + (wrow + 16 + l16) * HS + kb, khalf);
        v16h b  = loadB16(wd1 + n1 * HDIM + kb + ksel);
        acc0 = wmma(a0, b, acc0);
        acc1 = wmma(a1, b, acc1);
      }
      const float b1v = dbias[n1];
#pragma unroll
      for (int v = 0; v < 8; ++v) {
        const int M0 = wrow + v + hi16 * 8;
        const int M1 = M0 + 16;
        float u0 = acc0[v] + b1v;
        float u1 = acc1[v] + b1v;
        d1buf[M0 * D1S + n1] = (_Float16)((u0 > 0.0f) ? u0 : (__expf(u0) - 1.0f));
        d1buf[M1 * D1S + n1] = (_Float16)((u1 > 0.0f) ? u1 : (__expf(u1) - 1.0f));
      }
    }
    __syncthreads();

    // ---- decode layer 2 (K=64, cols 0..2 valid) + state update + output ----
    v8f a20 = {}, a21 = {};
#pragma unroll
    for (int kb2 = 0; kb2 < 2; ++kb2) {
      const int kb = kb2 * 32;
      v16h a0 = loadA16(&d1buf[(wrow +      l16) * D1S + kb], khalf);
      v16h a1 = loadA16(&d1buf[(wrow + 16 + l16) * D1S + kb], khalf);
      v16h b  = loadB16(wd2p + l16 * 64 + kb + ksel);
      a20 = wmma(a0, b, a20);
      a21 = wmma(a1, b, a21);
    }
    if (l16 < 3) {
      const float b2v = dbias[64 + l16];
#pragma unroll
      for (int v = 0; v < 8; ++v) {
        const int M0 = wrow + v + hi16 * 8;
        const int M1 = M0 + 16;
        float s0 = state_s[M0 * 4 + l16] + a20[v] + b2v;
        float s1 = state_s[M1 * 4 + l16] + a21[v] + b2v;
        state_s[M0 * 4 + l16] = s0;
        state_s[M1 * 4 + l16] = s1;
        out[(bbase + M0) * (T_STEPS * 3) + t * 3 + l16] = s0;
        out[(bbase + M1) * (T_STEPS * 3) + t * 3 + l16] = s1;
      }
    }
    ping = 1 - ping;
    __syncthreads();
  }
}

// ---------------------------------------------------------------------------
extern "C" void kernel_launch(void* const* d_in, const int* in_sizes, int n_in,
                              void* d_out, int out_size, void* d_ws, size_t ws_size,
                              hipStream_t stream) {
  (void)in_sizes; (void)n_in; (void)out_size; (void)ws_size;
  // setup_inputs() order:
  const float* init_hidden = (const float*)d_in[0];
  const float* plan        = (const float*)d_in[1];
  const float* gate        = (const float*)d_in[2];
  const float* init_state  = (const float*)d_in[3];
  const float* Wp   = (const float*)d_in[4];
  const float* bp   = (const float*)d_in[5];
  const float* Ws   = (const float*)d_in[6];
  const float* bs   = (const float*)d_in[7];
  const float* W_ih = (const float*)d_in[8];
  const float* b_ih = (const float*)d_in[9];
  const float* W_hh = (const float*)d_in[10];
  const float* b_hh = (const float*)d_in[11];
  const float* Wd1  = (const float*)d_in[12];
  const float* bd1  = (const float*)d_in[13];
  const float* Wd2  = (const float*)d_in[14];
  const float* bd2  = (const float*)d_in[15];
  float* out = (float*)d_out;

  _Float16* wsb = (_Float16*)d_ws;

  // 1) f32 -> f16 weight conversion (1152*384 is the largest array)
  int nconv = 1152 * 384;
  convert_weights_kernel<<<(nconv + 255) / 256, 256, 0, stream>>>(
      W_ih, W_hh, Wd1, Wd2, wsb);

  // 2) persistent GRU recurrence
  decoder_kernel<<<B_TOT / MT, 64, 0, stream>>>(
      init_hidden, plan, gate, init_state,
      Wp, bp, Ws, bs, b_ih, b_hh, bd1, bd2,
      wsb + OFF_WIH, wsb + OFF_WHH, wsb + OFF_WD1, wsb + OFF_WD2,
      out);
}
